// PerceiverAttention_36584531427506
// MI455X (gfx1250) — matrix operand order repaired and checked by reading, not compiled, not run on hardware
//
#include <hip/hip_runtime.h>
#include <hip/hip_bf16.h>

typedef __bf16 bf16_t;
typedef bf16_t v16bf __attribute__((ext_vector_type(16)));
typedef bf16_t v8bf  __attribute__((ext_vector_type(8)));
typedef bf16_t v4bf  __attribute__((ext_vector_type(4)));
typedef float  v8f   __attribute__((ext_vector_type(8)));
typedef float  v4f   __attribute__((ext_vector_type(4)));

#define DIMX   1024
#define HEADS  16
#define DH     64
#define BATCH  8
#define SEQN   4096
#define SEQM   64
#define NM     4160              // SEQN + SEQM
#define ROWS_KV (BATCH*NM)       // 33280
#define ROWS_Q  (BATCH*SEQM)     // 512
#define SPLITS 5
#define JB_PER_SPLIT 13          // 5*13*64 = 4160

// ---------- WMMA helpers (CDNA5 wave32) ----------

static __device__ __forceinline__ v16bf ld_pair(const bf16_t* p0, const bf16_t* p1) {
  v8bf a = *(const v8bf*)p0;
  v8bf b = *(const v8bf*)p1;
  return __builtin_shufflevector(a, b, 0,1,2,3,4,5,6,7,8,9,10,11,12,13,14,15);
}

static __device__ __forceinline__ v8f wmma_bf16(v16bf a, v16bf b, v8f c) {
  // (neg_a, A, neg_b, B, c_mod, C, reuse_a, reuse_b)
  return __builtin_amdgcn_wmma_f32_16x16x32_bf16(false, a, false, b, (short)0, c, false, false);
}

static __device__ __forceinline__ v8f zero8() {
  v8f z = {0.f,0.f,0.f,0.f,0.f,0.f,0.f,0.f};
  return z;
}

// ---------- Kernel 1: LayerNorm + cast to bf16 ----------
__global__ __launch_bounds__(256)
void ln_cast_kernel(const float* __restrict__ x, const float* __restrict__ lat,
                    const float* __restrict__ gm, const float* __restrict__ bm,
                    const float* __restrict__ gl, const float* __restrict__ bl,
                    bf16_t* __restrict__ kvin, bf16_t* __restrict__ latb) {
  int blk = blockIdx.x;              // 0 .. ROWS_KV-1
  int b = blk / NM;
  int r = blk % NM;
  bool is_lat = (r >= SEQN);
  const float* src; const float* g; const float* bb;
  if (!is_lat) { src = x   + ((size_t)b*SEQN + r)*DIMX;        g = gm; bb = bm; }
  else         { src = lat + ((size_t)b*SEQM + (r-SEQN))*DIMX; g = gl; bb = bl; }

  int tid = threadIdx.x;
  v4f v = ((const v4f*)src)[tid];
  float s  = v.x + v.y + v.z + v.w;
  float ss = v.x*v.x + v.y*v.y + v.z*v.z + v.w*v.w;
  #pragma unroll
  for (int off = 16; off > 0; off >>= 1) {
    s  += __shfl_down(s,  off);
    ss += __shfl_down(ss, off);
  }
  __shared__ float sw[8], ssw[8], mv[2];
  int wv = tid >> 5, ln = tid & 31;
  if (ln == 0) { sw[wv] = s; ssw[wv] = ss; }
  __syncthreads();
  if (tid == 0) {
    float ts = 0.f, tss = 0.f;
    #pragma unroll
    for (int i = 0; i < 8; ++i) { ts += sw[i]; tss += ssw[i]; }
    float mu  = ts * (1.0f/DIMX);
    float var = tss * (1.0f/DIMX) - mu*mu;
    mv[0] = mu; mv[1] = rsqrtf(var + 1e-5f);
  }
  __syncthreads();
  float mu = mv[0], rstd = mv[1];
  v4f gg = ((const v4f*)g)[tid];
  v4f bv = ((const v4f*)bb)[tid];
  v4bf o;
  o[0] = (bf16_t)((v.x - mu)*rstd*gg.x + bv.x);
  o[1] = (bf16_t)((v.y - mu)*rstd*gg.y + bv.y);
  o[2] = (bf16_t)((v.z - mu)*rstd*gg.z + bv.z);
  o[3] = (bf16_t)((v.w - mu)*rstd*gg.w + bv.w);
  ((v4bf*)(kvin + (size_t)blk*DIMX))[tid] = o;
  if (is_lat)
    ((v4bf*)(latb + ((size_t)b*SEQM + (r-SEQN))*DIMX))[tid] = o;
}

// ---------- Kernel 2: transpose + cast weight f32[K][N] -> bf16 WT[N][K] ----------
__global__ __launch_bounds__(256)
void transpose_cast_kernel(const float* __restrict__ W, bf16_t* __restrict__ WT,
                           int Krows, int Ncols) {
  size_t idx = (size_t)blockIdx.x*blockDim.x + threadIdx.x;
  size_t k = idx / (size_t)Ncols;
  size_t n = idx % (size_t)Ncols;
  WT[n*(size_t)Krows + k] = (bf16_t)W[idx];
}

// ---------- Kernel 3: WMMA GEMM  C[M][N] = A[M][K] * BT[N][K]^T ----------
// 256 threads = 8 waves; block tile 128x128; wave tile 32x64 (2 m x 4 n tiles).
// Ping-pong double-buffered fragments, unrolled by 2 so no register rotation
// copies are needed; prefetch 4 K-chunks ahead (global_prefetch).
// mode 0: bf16 row-major out. mode 1: f32 row-major out.
// mode 2: KV split: col<1024 -> K bf16 row-major; col>=1024 -> V^T bf16 packed.
__global__ __launch_bounds__(256)
void gemm_wmma(const bf16_t* __restrict__ A, const bf16_t* __restrict__ BT,
               void* __restrict__ Out, int Mrows, int Ncols, int Kdim,
               int mode, bf16_t* __restrict__ OutV) {
  int tid  = threadIdx.x;
  int wave = tid >> 5, lane = tid & 31;
  int L16 = lane & 15, hi = lane >> 4;
  int wm = wave >> 1, wn = wave & 1;
  int rowbase = blockIdx.y*128 + wm*32;
  int colbase = blockIdx.x*128 + wn*64;

  const bf16_t* arow0 = A + (size_t)(rowbase + L16)*Kdim + 8*hi;
  const bf16_t* arow1 = A + (size_t)(rowbase + 16 + L16)*Kdim + 8*hi;
  const bf16_t* brow[4];
  #pragma unroll
  for (int nt = 0; nt < 4; ++nt)
    brow[nt] = BT + (size_t)(colbase + nt*16 + L16)*Kdim + 8*hi;

  v8f acc[2][4];
  #pragma unroll
  for (int mt = 0; mt < 2; ++mt)
    #pragma unroll
    for (int nt = 0; nt < 4; ++nt)
      acc[mt][nt] = zero8();

  int kiters = Kdim >> 5;      // always even (>=4) for our shapes
  v16bf a0[2], b0[4], a1[2], b1[4];

  // prologue: chunks 0 and 1
  a0[0] = ld_pair(arow0, arow0 + 16);
  a0[1] = ld_pair(arow1, arow1 + 16);
  #pragma unroll
  for (int nt = 0; nt < 4; ++nt) b0[nt] = ld_pair(brow[nt], brow[nt] + 16);
  a1[0] = ld_pair(arow0 + 32, arow0 + 48);
  a1[1] = ld_pair(arow1 + 32, arow1 + 48);
  #pragma unroll
  for (int nt = 0; nt < 4; ++nt) b1[nt] = ld_pair(brow[nt] + 32, brow[nt] + 48);

  for (int kc = 0; kc + 2 < kiters; kc += 2) {
    int kb2 = (kc + 2) * 32;
    int kb3 = (kc + 3) * 32;
    // prefetch 4 chunks ahead (speculative; OOB prefetch is dropped)
    __builtin_prefetch(arow0 + kb2 + 64, 0, 3);
    __builtin_prefetch(arow1 + kb2 + 64, 0, 3);
    #pragma unroll
    for (int nt = 0; nt < 4; ++nt)
      __builtin_prefetch(brow[nt] + kb2 + 64, 0, 3);
    // consume buffer 0 (chunk kc), then refill it with chunk kc+2
    #pragma unroll
    for (int mt = 0; mt < 2; ++mt)
      #pragma unroll
      for (int nt = 0; nt < 4; ++nt)
        acc[mt][nt] = wmma_bf16(a0[mt], b0[nt], acc[mt][nt]);
    a0[0] = ld_pair(arow0 + kb2, arow0 + kb2 + 16);
    a0[1] = ld_pair(arow1 + kb2, arow1 + kb2 + 16);
    #pragma unroll
    for (int nt = 0; nt < 4; ++nt)
      b0[nt] = ld_pair(brow[nt] + kb2, brow[nt] + kb2 + 16);
    // consume buffer 1 (chunk kc+1), then refill it with chunk kc+3
    #pragma unroll
    for (int mt = 0; mt < 2; ++mt)
      #pragma unroll
      for (int nt = 0; nt < 4; ++nt)
        acc[mt][nt] = wmma_bf16(a1[mt], b1[nt], acc[mt][nt]);
    a1[0] = ld_pair(arow0 + kb3, arow0 + kb3 + 16);
    a1[1] = ld_pair(arow1 + kb3, arow1 + kb3 + 16);
    #pragma unroll
    for (int nt = 0; nt < 4; ++nt)
      b1[nt] = ld_pair(brow[nt] + kb3, brow[nt] + kb3 + 16);
  }
  // epilogue: last two chunks
  #pragma unroll
  for (int mt = 0; mt < 2; ++mt)
    #pragma unroll
    for (int nt = 0; nt < 4; ++nt)
      acc[mt][nt] = wmma_bf16(a0[mt], b0[nt], acc[mt][nt]);
  #pragma unroll
  for (int mt = 0; mt < 2; ++mt)
    #pragma unroll
    for (int nt = 0; nt < 4; ++nt)
      acc[mt][nt] = wmma_bf16(a1[mt], b1[nt], acc[mt][nt]);

  if (mode == 0) {
    bf16_t* O = (bf16_t*)Out;
    #pragma unroll
    for (int mt = 0; mt < 2; ++mt)
      #pragma unroll
      for (int nt = 0; nt < 4; ++nt) {
        int col  = colbase + nt*16 + L16;
        int row0 = rowbase + mt*16 + 8*hi;
        #pragma unroll
        for (int r = 0; r < 8; ++r)
          O[(size_t)(row0 + r)*Ncols + col] = (bf16_t)acc[mt][nt][r];
      }
  } else if (mode == 1) {
    float* O = (float*)Out;
    #pragma unroll
    for (int mt = 0; mt < 2; ++mt)
      #pragma unroll
      for (int nt = 0; nt < 4; ++nt) {
        int col  = colbase + nt*16 + L16;
        int row0 = rowbase + mt*16 + 8*hi;
        #pragma unroll
        for (int r = 0; r < 8; ++r)
          O[(size_t)(row0 + r)*Ncols + col] = acc[mt][nt][r];
      }
  } else {
    bf16_t* Kb = (bf16_t*)Out;
    #pragma unroll
    for (int mt = 0; mt < 2; ++mt)
      #pragma unroll
      for (int nt = 0; nt < 4; ++nt) {
        int col  = colbase + nt*16 + L16;
        int row0 = rowbase + mt*16 + 8*hi;
        if (col < DIMX) {
          #pragma unroll
          for (int r = 0; r < 8; ++r)
            Kb[(size_t)(row0 + r)*DIMX + col] = (bf16_t)acc[mt][nt][r];
        } else {
          int c2 = col - DIMX;
          int h = c2 >> 6, d = c2 & 63;
          int bb = row0 / NM, j0 = row0 % NM;   // 8-aligned run never crosses NM
          v8bf pk;
          #pragma unroll
          for (int r = 0; r < 8; ++r) pk[r] = (bf16_t)acc[mt][nt][r];
          *(v8bf*)(OutV + ((size_t)(bb*HEADS + h)*DH + d)*NM + j0) = pk;
        }
      }
  }
}

// ---------- Kernel 4: flash attention, split-j (flash-decoding) ----------
// Grid (128, SPLITS): block (bh, s) processes j in [s*832, s*832+832).
// Writes unnormalized partial o^T (f32) + per-row (m, l). 640 blocks x 4 waves.
__global__ __launch_bounds__(128)
void attn_kernel(const bf16_t* __restrict__ qb, const bf16_t* __restrict__ kb,
                 const bf16_t* __restrict__ vt,
                 float* __restrict__ po, float* __restrict__ pm,
                 float* __restrict__ pl) {
  __shared__ bf16_t P[4][16][72];   // per-wave P tile, padded rows

  int bh = blockIdx.x;              // 0..127
  int sp = blockIdx.y;              // 0..SPLITS-1
  int b = bh >> 4, h = bh & 15;
  int tid = threadIdx.x;
  int wave = tid >> 5, lane = tid & 31;
  int L16 = lane & 15, hi = lane >> 4;
  int iq = wave*16 + L16;           // query row in [0,64)

  size_t qoff = ((size_t)b*SEQM + iq)*DIMX + h*DH;
  v16bf qf[2];
  #pragma unroll
  for (int kc = 0; kc < 2; ++kc) {
    const bf16_t* p = qb + qoff + kc*32 + 8*hi;
    qf[kc] = ld_pair(p, p + 16);
  }

  const float sc2 = 0.125f * 1.44269504088896340736f; // dh^-0.5 * log2(e)
  float m_i = -__builtin_inff();
  float l_i = 0.0f;
  v8f o[4];
  #pragma unroll
  for (int dt = 0; dt < 4; ++dt) o[dt] = zero8();

  const bf16_t* kbase = kb + (size_t)b*NM*DIMX + h*DH;
  const bf16_t* vbase = vt + (size_t)(b*HEADS + h)*DH*NM;

  int jb0 = sp * JB_PER_SPLIT;
  for (int jb = jb0; jb < jb0 + JB_PER_SPLIT; ++jb) {
    int j0 = jb*64;
    // ---- dotsT[j][i] = sum_d K[j,d] * q[i,d]
    v8f s[4];
    #pragma unroll
    for (int jt = 0; jt < 4; ++jt) s[jt] = zero8();
    #pragma unroll
    for (int jt = 0; jt < 4; ++jt) {
      const bf16_t* kp = kbase + (size_t)(j0 + jt*16 + L16)*DIMX;
      #pragma unroll
      for (int kc = 0; kc < 2; ++kc) {
        v16bf kf = ld_pair(kp + kc*32 + 8*hi, kp + kc*32 + 16 + 8*hi);
        s[jt] = wmma_bf16(kf, qf[kc], s[jt]);
      }
    }
    // ---- online softmax over j (VGPR dim + half-wave swap)
    float mn = -__builtin_inff();
    #pragma unroll
    for (int jt = 0; jt < 4; ++jt)
      #pragma unroll
      for (int r = 0; r < 8; ++r) {
        s[jt][r] *= sc2;
        mn = fmaxf(mn, s[jt][r]);
      }
    mn = fmaxf(mn, __shfl_xor(mn, 16));
    float mnew  = fmaxf(m_i, mn);
    float alpha = exp2f(m_i - mnew);
    m_i = mnew;
    float rs = 0.0f;
    #pragma unroll
    for (int jt = 0; jt < 4; ++jt) {
      v8bf pk;
      #pragma unroll
      for (int r = 0; r < 8; ++r) {
        float e = exp2f(s[jt][r] - m_i);
        rs += e;
        pk[r] = (bf16_t)e;
      }
      *(v8bf*)(&P[wave][L16][jt*16 + 8*hi]) = pk;   // P row-major [i][j]
    }
    rs += __shfl_xor(rs, 16);
    l_i = l_i*alpha + rs;
    #pragma unroll
    for (int dt = 0; dt < 4; ++dt)
      #pragma unroll
      for (int r = 0; r < 8; ++r) o[dt][r] *= alpha;
    // ---- oT[d][i] += sum_j V^T[d,j] * P[i,j]
    #pragma unroll
    for (int dt = 0; dt < 4; ++dt) {
      const bf16_t* vp = vbase + (size_t)(dt*16 + L16)*NM + j0;
      #pragma unroll
      for (int kc = 0; kc < 2; ++kc) {
        v16bf vf = ld_pair(vp + kc*32 + 16*hi, vp + kc*32 + 8 + 16*hi);
        const bf16_t* pp = &P[wave][L16][kc*32 + 16*hi];
        v16bf pf = ld_pair(pp, pp + 8);
        o[dt] = wmma_bf16(vf, pf, o[dt]);
      }
    }
  }

  // partial store: po[((sp*128+bh)*64 + i)*64 + d], unnormalized
  size_t pbase = ((size_t)(sp*128 + bh)*64 + iq)*64;
  #pragma unroll
  for (int dt = 0; dt < 4; ++dt) {
    *(v8f*)(po + pbase + dt*16 + 8*hi) = o[dt];
  }
  if (hi == 0) {
    size_t mi = (size_t)(sp*128 + bh)*64 + iq;
    pm[mi] = m_i;
    pl[mi] = l_i;
  }
}

// ---------- Kernel 5: flash-decoding combine -> bf16 attention output ----------
// One block per (b,h); thread t handles row i = t/4, d-chunk (t%4)*16.
__global__ __launch_bounds__(256)
void attn_combine(const float* __restrict__ po, const float* __restrict__ pm,
                  const float* __restrict__ pl, bf16_t* __restrict__ ob) {
  int bh = blockIdx.x;              // 0..127
  int b = bh >> 4, h = bh & 15;
  int t = threadIdx.x;
  int i = t >> 2;
  int dc = (t & 3) * 16;

  float mv[SPLITS];
  float M = -__builtin_inff();
  #pragma unroll
  for (int s = 0; s < SPLITS; ++s) {
    mv[s] = pm[(size_t)(s*128 + bh)*64 + i];
    M = fmaxf(M, mv[s]);
  }
  float w[SPLITS];
  float L = 0.0f;
  #pragma unroll
  for (int s = 0; s < SPLITS; ++s) {
    w[s] = exp2f(mv[s] - M);
    L += w[s] * pl[(size_t)(s*128 + bh)*64 + i];
  }
  float invL = 1.0f / L;

  float acc[16];
  #pragma unroll
  for (int e = 0; e < 16; ++e) acc[e] = 0.0f;
  for (int s = 0; s < SPLITS; ++s) {
    const float* p = po + ((size_t)(s*128 + bh)*64 + i)*64 + dc;
    #pragma unroll
    for (int c = 0; c < 4; ++c) {
      v4f v = ((const v4f*)p)[c];
      acc[c*4+0] += w[s]*v.x; acc[c*4+1] += w[s]*v.y;
      acc[c*4+2] += w[s]*v.z; acc[c*4+3] += w[s]*v.w;
    }
  }
  bf16_t* op = ob + ((size_t)b*SEQM + i)*DIMX + h*DH + dc;
  #pragma unroll
  for (int c = 0; c < 2; ++c) {
    v8bf pk;
    #pragma unroll
    for (int e = 0; e < 8; ++e) pk[e] = (bf16_t)(acc[c*8+e]*invL);
    ((v8bf*)op)[c] = pk;
  }
}

// ---------- host launcher ----------
extern "C" void kernel_launch(void* const* d_in, const int* in_sizes, int n_in,
                              void* d_out, int out_size, void* d_ws, size_t ws_size,
                              hipStream_t stream) {
  const float* x    = (const float*)d_in[0];
  const float* lat  = (const float*)d_in[1];
  const float* gm   = (const float*)d_in[2];
  const float* bm   = (const float*)d_in[3];
  const float* gl   = (const float*)d_in[4];
  const float* bl   = (const float*)d_in[5];
  const float* Wq   = (const float*)d_in[6];
  const float* Wkv  = (const float*)d_in[7];
  const float* Wout = (const float*)d_in[8];
  float* out = (float*)d_out;
  (void)in_sizes; (void)n_in; (void)out_size; (void)ws_size;

  char* ws = (char*)d_ws;
  size_t off = 0;
  auto alloc = [&](size_t bytes) -> char* {
    char* p = ws + off;
    off += (bytes + 255) & ~(size_t)255;
    return p;
  };
  bf16_t* kvin  = (bf16_t*)alloc((size_t)ROWS_KV*DIMX*2);        // LN(concat(x,lat))
  bf16_t* latb  = (bf16_t*)alloc((size_t)ROWS_Q*DIMX*2);         // LN(lat) contiguous
  bf16_t* WqT   = (bf16_t*)alloc((size_t)DIMX*DIMX*2);
  bf16_t* WkvT  = (bf16_t*)alloc((size_t)2*DIMX*DIMX*2);
  bf16_t* WoutT = (bf16_t*)alloc((size_t)DIMX*DIMX*2);
  bf16_t* qbuf  = (bf16_t*)alloc((size_t)ROWS_Q*DIMX*2);         // q row-major
  bf16_t* kbuf  = (bf16_t*)alloc((size_t)ROWS_KV*DIMX*2);        // K row-major
  bf16_t* vtbuf = (bf16_t*)alloc((size_t)BATCH*HEADS*DH*NM*2);   // V transposed
  bf16_t* attnb = (bf16_t*)alloc((size_t)ROWS_Q*DIMX*2);         // attn out bf16
  float*  po    = (float*)alloc((size_t)SPLITS*128*64*64*4);     // partial o^T
  float*  pmb   = (float*)alloc((size_t)SPLITS*128*64*4);        // partial max
  float*  plb   = (float*)alloc((size_t)SPLITS*128*64*4);        // partial sum

  ln_cast_kernel<<<ROWS_KV, 256, 0, stream>>>(x, lat, gm, bm, gl, bl, kvin, latb);
  transpose_cast_kernel<<<(DIMX*DIMX)/256,   256, 0, stream>>>(Wq,   WqT,   DIMX, DIMX);
  transpose_cast_kernel<<<(2*DIMX*DIMX)/256, 256, 0, stream>>>(Wkv,  WkvT,  DIMX, 2*DIMX);
  transpose_cast_kernel<<<(DIMX*DIMX)/256,   256, 0, stream>>>(Wout, WoutT, DIMX, DIMX);

  // q = LN(lat) @ Wq  -> bf16 row-major [512][1024]
  gemm_wmma<<<dim3(DIMX/128, ROWS_Q/128), 256, 0, stream>>>(
      latb, WqT, qbuf, ROWS_Q, DIMX, DIMX, 0, nullptr);
  // kv = kvin @ Wkv -> K row-major + V transposed (dominant GEMM, 139.6 GFLOP)
  gemm_wmma<<<dim3(2*DIMX/128, ROWS_KV/128), 256, 0, stream>>>(
      kvin, WkvT, kbuf, ROWS_KV, 2*DIMX, DIMX, 2, vtbuf);
  // flash attention, split-j for occupancy (640 blocks)
  attn_kernel<<<dim3(BATCH*HEADS, SPLITS), 128, 0, stream>>>(
      qbuf, kbuf, vtbuf, po, pmb, plb);
  attn_combine<<<BATCH*HEADS, 256, 0, stream>>>(po, pmb, plb, attnb);
  // out = attn @ Wout -> f32
  gemm_wmma<<<dim3(DIMX/128, ROWS_Q/128), 256, 0, stream>>>(
      attnb, WoutT, out, ROWS_Q, DIMX, DIMX, 1, nullptr);
}
